// GNNWithEmbeddings_19155554140461
// MI455X (gfx1250) — compile-verified, matrix operand order checked by reading, not run on hardware
//
#include <hip/hip_runtime.h>
#include <hip/hip_bf16.h>

// ---------------------------------------------------------------------------
// GraphSAGE 2-layer + classifier for MI455X (gfx1250, wave32, WMMA).
//   N=50000 nodes, E=800000 edges, 128 -> 256 -> 256 -> 40 channels.
// Pipeline:
//   1) deg[dst] += 1                      (f32 hw atomics, L2-resident)
//   2) agg_sum[dst] += x[src]             (f32 hw atomics, vectorized loads)
//   3) invdeg = 1/max(deg,1)
//   4) h = relu((agg_sum*invdeg) @ w_l^T + b + x @ w_r^T)   [bf16 WMMA]
//   5) repeat 2..4 on h for layer 2
//   6) out = z @ wc^T + bc                                  [bf16 WMMA]
// GEMM: one wave owns a 16x64 output tile (4 n-tiles); A fragment loaded once
// per K-chunk and fed to 4 independent WMMAs. Template-specialized so the
// full-tile layers have a branch-free inner loop (no exec masking).
// ---------------------------------------------------------------------------

typedef __attribute__((ext_vector_type(16))) __bf16 v16bf;
typedef __attribute__((ext_vector_type(8)))  float  v8f;

#define N_NODES 50000
#define N_EDGES 800000
#define IN_CH   128
#define HID_CH  256
#define OUT_CH  40
#define NTB     4        // n-tiles (of 16) per wave -> 16x64 output per wave

// ---------------------------------------------------------------------------
// Edge kernels
// ---------------------------------------------------------------------------

__global__ void edge_degree_kernel(const long long* __restrict__ ei,
                                   float* __restrict__ deg, int E) {
  int e = blockIdx.x * blockDim.x + threadIdx.x;
  if (e < E) {
    int d = (int)ei[(long long)E + e];     // dst row of edge_index
    unsafeAtomicAdd(&deg[d], 1.0f);        // -> global_atomic_add_f32
  }
}

// agg_sum[dst[e], :] += X[src[e], :]
// One thread per (edge, 4-channel group): lanes of a wave share one edge ->
// coalesced float4 gather of the source row, 4 f32 atomics to the dst row.
__global__ void edge_scatter_add_kernel(const long long* __restrict__ ei,
                                        const float* __restrict__ X,
                                        float* __restrict__ S,
                                        int E, int C) {
  long long tid = (long long)blockIdx.x * blockDim.x + threadIdx.x;
  int groups = C >> 2;
  long long total = (long long)E * groups;
  if (tid >= total) return;
  int e = (int)(tid / groups);
  int g = (int)(tid - (long long)e * groups);
  int s = (int)ei[e];                      // src
  int d = (int)ei[(long long)E + e];       // dst
  const float4 v = *(const float4*)(X + (size_t)s * C + (size_t)g * 4);
  float* p = S + (size_t)d * C + (size_t)g * 4;
  unsafeAtomicAdd(p + 0, v.x);
  unsafeAtomicAdd(p + 1, v.y);
  unsafeAtomicAdd(p + 2, v.z);
  unsafeAtomicAdd(p + 3, v.w);
}

__global__ void invdeg_kernel(float* __restrict__ deg, int n) {
  int i = blockIdx.x * blockDim.x + threadIdx.x;
  if (i < n) deg[i] = 1.0f / fmaxf(deg[i], 1.0f);
}

// ---------------------------------------------------------------------------
// bf16 WMMA fragment loader (CDNA5 16x16x32 layout, wave32)
//   A (16x32, MxK): lane l -> row l&15; lanes 0-15 hold K {0..7,16..23},
//                   lanes 16-31 hold K {8..15,24..31} (2 bf16 per VGPR).
//   B (32x16, KxN): identical pattern with lane -> column n = l&15.
//   Loading w[n][k] row-major as the B fragment realizes x @ w^T.
// rowp points at element (row, kc + khalf); stride within row is 1.
// ---------------------------------------------------------------------------
__device__ inline v16bf load_frag_bf16(const float* __restrict__ rowp, float scale) {
  float4 a = *(const float4*)(rowp + 0);
  float4 b = *(const float4*)(rowp + 4);
  float4 c = *(const float4*)(rowp + 16);
  float4 d = *(const float4*)(rowp + 20);
  float f[16] = {a.x, a.y, a.z, a.w, b.x, b.y, b.z, b.w,
                 c.x, c.y, c.z, c.w, d.x, d.y, d.z, d.w};
  v16bf r;
#pragma unroll
  for (int i = 0; i < 16; ++i) r[i] = (__bf16)(f[i] * scale);
  return r;
}

// ---------------------------------------------------------------------------
// Fused SAGE GEMM, 16x64 tile per wave:
//   Out[m,n] = act( bias[n] + sum_k (A1[m,k]*invdeg[m]) * W1[n,k]
//                           [+ sum_k  A2[m,k]           * W2[n,k]] )
// Template parameters:
//   K         compile-time inner dimension (128 or 256)
//   BOUNDED   true only for the classifier (OUTC=40 not a multiple of 64)
//   HAS_SELF  second matmul term present (SAGE layers)
//   HAS_SCALE mean-aggregation row scaling present
//   RELU      apply relu on store
// blockDim = (32, 4). EXEC is full at every WMMA (wave-uniform early exit).
// ---------------------------------------------------------------------------
template <int K, bool BOUNDED, bool HAS_SELF, bool HAS_SCALE, bool RELU>
__global__ void sage_wmma_gemm(const float* __restrict__ A1,
                               const float* __restrict__ invdeg,
                               const float* __restrict__ W1,
                               const float* __restrict__ A2,
                               const float* __restrict__ W2,
                               const float* __restrict__ bias,
                               float* __restrict__ Out,
                               int OUTC, int Mt, int Ntb) {
  int lane = threadIdx.x;                       // 0..31
  int tileId = blockIdx.x * blockDim.y + threadIdx.y;
  if (tileId >= Mt * Ntb) return;               // uniform per wave
  int mt  = tileId / Ntb;
  int ntb = tileId - mt * Ntb;

  int hi  = lane >> 4;                          // 0 or 1 (K/row half select)
  int idx = lane & 15;
  int khalf = hi * 8;
  int m = mt * 16 + idx;                        // A row this lane feeds

  int  n[NTB];
  bool nv[NTB];
#pragma unroll
  for (int t = 0; t < NTB; ++t) {
    n[t]  = ntb * (16 * NTB) + t * 16 + idx;    // output column per sub-tile
    nv[t] = BOUNDED ? (n[t] < OUTC) : true;
  }

  v16bf zfrag;
#pragma unroll
  for (int i = 0; i < 16; ++i) zfrag[i] = (__bf16)0.0f;

  // Accumulators initialized with the bias (every element of lane l is col n[t])
  v8f acc[NTB];
#pragma unroll
  for (int t = 0; t < NTB; ++t) {
    float bv = nv[t] ? bias[n[t]] : 0.0f;
#pragma unroll
    for (int i = 0; i < 8; ++i) acc[t][i] = bv;
  }

  const float* wrow1[NTB];
#pragma unroll
  for (int t = 0; t < NTB; ++t)
    wrow1[t] = W1 + (size_t)(nv[t] ? n[t] : 0) * K + khalf;

  // Pass 1: (A1 * invdeg[m]) @ W1^T   -- mean aggregation folded into A load
  {
    float scale = HAS_SCALE ? invdeg[m] : 1.0f;
    const float* a1row = A1 + (size_t)m * K + khalf;
#pragma unroll 4
    for (int kc = 0; kc < K; kc += 32) {
      v16bf a = load_frag_bf16(a1row + kc, scale);
#pragma unroll
      for (int t = 0; t < NTB; ++t) {
        v16bf b;
        if (BOUNDED)
          b = nv[t] ? load_frag_bf16(wrow1[t] + kc, 1.0f) : zfrag;
        else
          b = load_frag_bf16(wrow1[t] + kc, 1.0f);
        acc[t] = __builtin_amdgcn_wmma_f32_16x16x32_bf16(false, a, false, b,
                                                         (short)0, acc[t],
                                                         false, false);
      }
    }
  }

  // Pass 2 (optional): A2 @ W2^T  (self/root term of SAGEConv)
  if (HAS_SELF) {
    const float* wrow2[NTB];
#pragma unroll
    for (int t = 0; t < NTB; ++t)
      wrow2[t] = W2 + (size_t)(nv[t] ? n[t] : 0) * K + khalf;
    const float* a2row = A2 + (size_t)m * K + khalf;
#pragma unroll 4
    for (int kc = 0; kc < K; kc += 32) {
      v16bf a = load_frag_bf16(a2row + kc, 1.0f);
#pragma unroll
      for (int t = 0; t < NTB; ++t) {
        v16bf b;
        if (BOUNDED)
          b = nv[t] ? load_frag_bf16(wrow2[t] + kc, 1.0f) : zfrag;
        else
          b = load_frag_bf16(wrow2[t] + kc, 1.0f);
        acc[t] = __builtin_amdgcn_wmma_f32_16x16x32_bf16(false, a, false, b,
                                                         (short)0, acc[t],
                                                         false, false);
      }
    }
  }

  // Store: C/D layout -> VGPR i holds row (mt*16 + hi*8 + i), col n[t].
  int rowbase = mt * 16 + hi * 8;
#pragma unroll
  for (int t = 0; t < NTB; ++t) {
    if (nv[t]) {
#pragma unroll
      for (int i = 0; i < 8; ++i) {
        float v = acc[t][i];
        if (RELU) v = fmaxf(v, 0.0f);
        Out[(size_t)(rowbase + i) * OUTC + n[t]] = v;
      }
    }
  }
}

// ---------------------------------------------------------------------------
// Host-side launch (graph-capture safe: only async stream ops)
// ---------------------------------------------------------------------------
extern "C" void kernel_launch(void* const* d_in, const int* in_sizes, int n_in,
                              void* d_out, int out_size, void* d_ws, size_t ws_size,
                              hipStream_t stream) {
  const float*     x   = (const float*)d_in[0];
  const long long* ei  = (const long long*)d_in[1];   // int64 per reference
  const float*     w1l = (const float*)d_in[2];
  const float*     b1l = (const float*)d_in[3];
  const float*     w1r = (const float*)d_in[4];
  const float*     w2l = (const float*)d_in[5];
  const float*     b2l = (const float*)d_in[6];
  const float*     w2r = (const float*)d_in[7];
  const float*     wc  = (const float*)d_in[8];
  const float*     bc  = (const float*)d_in[9];
  float* out = (float*)d_out;

  const int N = N_NODES, E = N_EDGES;

  // Workspace layout (floats): deg | bufA (agg_sum, 256ch) | bufB (h) | bufC (z)
  float* deg  = (float*)d_ws;
  float* bufA = deg  + 50176;                  // N padded
  float* bufB = bufA + (size_t)N * HID_CH;
  float* bufC = bufB + (size_t)N * HID_CH;

  // ---- Layer 1 aggregation ----
  hipMemsetAsync(deg,  0, 50176 * sizeof(float), stream);
  hipMemsetAsync(bufA, 0, (size_t)N * IN_CH * sizeof(float), stream);

  edge_degree_kernel<<<(E + 255) / 256, 256, 0, stream>>>(ei, deg, E);
  {
    long long total = (long long)E * (IN_CH / 4);
    edge_scatter_add_kernel<<<(unsigned)((total + 255) / 256), 256, 0, stream>>>(
        ei, x, bufA, E, IN_CH);
  }
  invdeg_kernel<<<(N + 255) / 256, 256, 0, stream>>>(deg, N);

  // ---- Layer 1 GEMM: h = relu(agg @ w1_l^T + b1_l + x @ w1_r^T) ----
  dim3 blk(32, 4);
  int Mt = N / 16;                              // 3125 exact
  {
    int Ntb = HID_CH / (16 * NTB);              // 4 blocks of 64 cols
    int tiles = Mt * Ntb;
    sage_wmma_gemm<IN_CH, false, true, true, true>
        <<<(tiles + 3) / 4, blk, 0, stream>>>(
            bufA, deg, w1l, x, w1r, b1l, bufB, HID_CH, Mt, Ntb);
  }

  // ---- Layer 2 aggregation (same graph, reuse invdeg) ----
  hipMemsetAsync(bufA, 0, (size_t)N * HID_CH * sizeof(float), stream);
  {
    long long total = (long long)E * (HID_CH / 4);
    edge_scatter_add_kernel<<<(unsigned)((total + 255) / 256), 256, 0, stream>>>(
        ei, bufB, bufA, E, HID_CH);
  }

  // ---- Layer 2 GEMM: z = relu(agg @ w2_l^T + b2_l + h @ w2_r^T) ----
  {
    int Ntb = HID_CH / (16 * NTB);
    int tiles = Mt * Ntb;
    sage_wmma_gemm<HID_CH, false, true, true, true>
        <<<(tiles + 3) / 4, blk, 0, stream>>>(
            bufA, deg, w2l, bufB, w2r, b2l, bufC, HID_CH, Mt, Ntb);
  }

  // ---- Classifier: out = z @ wc^T + bc  (OUTC=40 < 64 -> masked tiles) ----
  {
    int Ntb = (OUT_CH + 16 * NTB - 1) / (16 * NTB);   // 1
    int tiles = Mt * Ntb;
    sage_wmma_gemm<HID_CH, true, false, false, false>
        <<<(tiles + 3) / 4, blk, 0, stream>>>(
            bufC, nullptr, wc, nullptr, nullptr, bc, out, OUT_CH, Mt, Ntb);
  }
}
